// QuantumNeuralNetwork_51599737094575
// MI455X (gfx1250) — matrix operand order repaired and checked by reading, not compiled, not run on hardware
//
#include <hip/hip_runtime.h>
#include <hip/hip_bf16.h>

typedef _Float16 v16h __attribute__((ext_vector_type(16)));
typedef _Float16 v8h  __attribute__((ext_vector_type(8)));
typedef float    v8f  __attribute__((ext_vector_type(8)));
typedef unsigned int v4u __attribute__((ext_vector_type(4)));
typedef int      v4i __attribute__((ext_vector_type(4)));
typedef int      v8i __attribute__((ext_vector_type(8)));

constexpr int BROWS = 65536;
constexpr int D_IN  = 1024;
constexpr int NQ    = 64;
constexpr int D_OUT = 1024;
constexpr float INV_2PI = 0.15915494309189535f;   // v_cos/v_sin take revolutions

// fast trig for small |x| (quantum params ~ N(0,0.1): no range reduction needed)
__device__ __forceinline__ float fcos(float x) {
    return __builtin_amdgcn_cosf(x * INV_2PI);
}
__device__ __forceinline__ float fsin(float x) {
    return __builtin_amdgcn_sinf(x * INV_2PI);
}

// ---------------------------------------------------------------------------
// Prolog 1: convert W_in (NQ x D_IN, f32) -> f16
// ---------------------------------------------------------------------------
__global__ __launch_bounds__(256)
void qnn_prep_win(const float* __restrict__ W_in, _Float16* __restrict__ Wi16) {
    int i = blockIdx.x * 256 + threadIdx.x;   // 65536 elements
    Wi16[i] = (_Float16)W_in[i];
}

// ---------------------------------------------------------------------------
// Prolog 2: fold quantum params into effective output weights / bias.
//   W_eff[d,j] = W_out[d,j] * cos(p0_j)        (f16)
//   b_eff[d]   = b_out[d] + sum_j sin(p1_j)*cos(p2_j)*W_out[d,j]
// ---------------------------------------------------------------------------
__global__ __launch_bounds__(256)
void qnn_prep_wout(const float* __restrict__ qp,
                   const float* __restrict__ W_out,
                   const float* __restrict__ b_out,
                   _Float16* __restrict__ We16,
                   float* __restrict__ b_eff) {
    int d = blockIdx.x * 256 + threadIdx.x;   // 1024 rows of W_out
    float acc = b_out[d];
    for (int j = 0; j < NQ; ++j) {
        float w  = W_out[d * NQ + j];
        float p0 = qp[j * 3 + 0];
        float p1 = qp[j * 3 + 1];
        float p2 = qp[j * 3 + 2];
        We16[d * NQ + j] = (_Float16)(w * fcos(p0));
        acc += fsin(p1) * fcos(p2) * w;
    }
    b_eff[d] = acc;
}

// branch-free: cos(pi/2 * tanh(v))
//   tanh(v) = 1 - 2/(1 + 2^(2v*log2(e)))    (exact saturation at both tails)
//   v_cos_f32 computes cos(2*pi*x)  =>  cos(pi/2 * t) = v_cos(t * 0.25)
__device__ __forceinline__ float cos_halfpi_tanh(float v) {
    float e  = __builtin_amdgcn_exp2f(v * 2.8853900817779268f);
    float th = 1.0f - 2.0f * __builtin_amdgcn_rcpf(1.0f + e);
    return __builtin_amdgcn_cosf(th * 0.25f);
}

// ---------------------------------------------------------------------------
// Fused main kernel: out = cos(pi/2 * tanh(x @ W_in^T + b_in)) @ W_eff^T + b_eff
// 256 threads = 8 wave32; each wave owns 16 rows; 128 rows per workgroup.
// W_in (f16, 128KB) staged in LDS via TDM; x streamed from HBM (roofline term).
// ---------------------------------------------------------------------------
__global__ __launch_bounds__(256)
void qnn_main(const float* __restrict__ x,
              const float* __restrict__ b_in,
              const _Float16* __restrict__ Wi16,
              const _Float16* __restrict__ We16,
              const float* __restrict__ b_eff,
              float* __restrict__ out) {
    // LDS: [0, 65536) halves = W_in staged; [65536, 73728) = per-wave s tiles
    __shared__ __align__(256) _Float16 smem[NQ * D_IN + 8 * 16 * NQ];
    _Float16* Wi  = smem;
    _Float16* s_w = smem + NQ * D_IN;

    const int tid  = threadIdx.x;
    const int wave = tid >> 5;
    const int lane = tid & 31;
    const int l15  = lane & 15;
    const int hi   = lane >> 4;                 // lane half: 0 or 1
    const int row0 = blockIdx.x * 128 + wave * 16;

    // ---- stage W_in f16 (64 x 1024) into LDS via Tensor Data Mover -------
#if __has_builtin(__builtin_amdgcn_tensor_load_to_lds) && __has_builtin(__builtin_amdgcn_s_wait_tensorcnt)
    if (wave == 0) {
        unsigned lds_base = (unsigned)(uintptr_t)&Wi[0];
        unsigned long long ga = (unsigned long long)(uintptr_t)Wi16;
        // D# group0: count=1 | lds_addr | global_addr[56:0] | type=2
        v4u g0;
        g0[0] = 1u;
        g0[1] = lds_base;
        g0[2] = (unsigned)(ga & 0xFFFFFFFFull);
        g0[3] = (unsigned)((ga >> 32) & 0x1FFFFFFull) | (2u << 30);
        // D# group1: data_size=2B; tensor 1024 x 64, tile 1024 x 64, stride 1024
        v8i g1;
        g1[0] = (1 << 16);            // wg_mask=0, data_size=1 (2 bytes)
        g1[1] = (int)(1024u << 16);   // [63:48] = tensor_dim0 low16 = 1024
        g1[2] = (int)(64u << 16);     // [79:64] = dim0 hi = 0; [95:80] = tensor_dim1 = 64
        g1[3] = (int)(1024u << 16);   // [111:96] = dim1 hi = 0; [127:112] = tile_dim0 = 1024
        g1[4] = 64;                   // [143:128] = tile_dim1 = 64; tile_dim2 = 0
        g1[5] = 1024;                 // tensor_dim0_stride low32 = 1024
        g1[6] = 0;
        g1[7] = 0;
        v4i gz = (v4i){0, 0, 0, 0};
#if defined(__clang_major__) && (__clang_major__ >= 23)
        v8i gz8 = (v8i){0, 0, 0, 0, 0, 0, 0, 0};
        __builtin_amdgcn_tensor_load_to_lds(g0, g1, gz, gz, gz8, 0);
#else
        __builtin_amdgcn_tensor_load_to_lds(g0, g1, gz, gz, 0);
#endif
        __builtin_amdgcn_s_wait_tensorcnt(0);
    }
#else
    for (int i = tid; i < (NQ * D_IN) / 8; i += 256)
        ((v8h*)Wi)[i] = ((const v8h*)Wi16)[i];
#endif
    __syncthreads();

    // ====================== GEMM1: h = x @ W_in^T ==========================
    v8f c[4];
#pragma unroll
    for (int t = 0; t < 4; ++t) c[t] = (v8f){};

    const float* xp = x + (size_t)(row0 + l15) * D_IN;

    for (int k0 = 0; k0 < D_IN; k0 += 32) {
        // A fragment (16x32 f16, contiguous K order):
        //   lanes 0-15 : K = [k0, k0+16)
        //   lanes 16-31: K = [k0+16, k0+32)
        const float* xc = xp + k0 + hi * 16;
        __builtin_prefetch(xc + 64, 0, 1);      // k-chunk two iterations ahead -> global_prefetch
        float4 f0 = *(const float4*)(xc);
        float4 f1 = *(const float4*)(xc + 4);
        float4 f2 = *(const float4*)(xc + 8);
        float4 f3 = *(const float4*)(xc + 12);
        v16h a;
        a[0]  = (_Float16)f0.x; a[1]  = (_Float16)f0.y;
        a[2]  = (_Float16)f0.z; a[3]  = (_Float16)f0.w;
        a[4]  = (_Float16)f1.x; a[5]  = (_Float16)f1.y;
        a[6]  = (_Float16)f1.z; a[7]  = (_Float16)f1.w;
        a[8]  = (_Float16)f2.x; a[9]  = (_Float16)f2.y;
        a[10] = (_Float16)f2.z; a[11] = (_Float16)f2.w;
        a[12] = (_Float16)f3.x; a[13] = (_Float16)f3.y;
        a[14] = (_Float16)f3.z; a[15] = (_Float16)f3.w;

#pragma unroll
        for (int t = 0; t < 4; ++t) {
            // B fragment from LDS: column n of B = row n of W_in (contiguous).
            const _Float16* wp = Wi + (t * 16 + l15) * D_IN + k0 + hi * 16;
            v16h b = *(const v16h*)wp;
            c[t] = __builtin_amdgcn_wmma_f32_16x16x32_f16(
                       false, a, false, b, (short)0, c[t], false, false);
        }
    }

    // ============ elementwise: s = cos(pi/2 * tanh(h + b_in)) ==============
    // C-layout: lane holds column (t*16 + l15), rows hi*8 .. hi*8+7.
    _Float16* sw = s_w + wave * 16 * NQ;
#pragma unroll
    for (int t = 0; t < 4; ++t) {
        const int col = t * 16 + l15;
        const float bi = b_in[col];
#pragma unroll
        for (int r = 0; r < 8; ++r) {
            const int m = hi * 8 + r;
            sw[m * NQ + col] = (_Float16)cos_halfpi_tanh(c[t][r] + bi);
        }
    }
    __syncthreads();   // lane->lane transpose through LDS

    // ====================== GEMM2: out = s @ W_eff^T =======================
    v8h p0 = *(const v8h*)(sw + l15 * NQ + hi * 16);
    v8h p1 = *(const v8h*)(sw + l15 * NQ + hi * 16 + 8);
    v8h p2 = *(const v8h*)(sw + l15 * NQ + hi * 16 + 32);
    v8h p3 = *(const v8h*)(sw + l15 * NQ + hi * 16 + 40);
    v16h a0 = __builtin_shufflevector(p0, p1, 0,1,2,3,4,5,6,7,8,9,10,11,12,13,14,15);
    v16h a1 = __builtin_shufflevector(p2, p3, 0,1,2,3,4,5,6,7,8,9,10,11,12,13,14,15);

    for (int nt = 0; nt < D_OUT / 16; ++nt) {
        const int col = nt * 16 + l15;
        // B fragments: column d of B = row d of W_eff (64 contiguous f16).
        const _Float16* wp = We16 + (size_t)col * NQ + hi * 16;
        v16h b0 = *(const v16h*)wp;          // K = 0..31
        v16h b1 = *(const v16h*)(wp + 32);   // K = 32..63
        v8f d = (v8f){};
        d = __builtin_amdgcn_wmma_f32_16x16x32_f16(
                false, a0, false, b0, (short)0, d, false, false);
        d = __builtin_amdgcn_wmma_f32_16x16x32_f16(
                false, a1, false, b1, (short)0, d, false, false);

        const float be = b_eff[col];
#pragma unroll
        for (int r = 0; r < 8; ++r) {
            out[(size_t)(row0 + hi * 8 + r) * D_OUT + col] = d[r] + be;
        }
    }
}

// ---------------------------------------------------------------------------
// Launch
// ---------------------------------------------------------------------------
extern "C" void kernel_launch(void* const* d_in, const int* in_sizes, int n_in,
                              void* d_out, int out_size, void* d_ws, size_t ws_size,
                              hipStream_t stream) {
    const float* x     = (const float*)d_in[0];
    const float* W_in  = (const float*)d_in[1];
    const float* b_in  = (const float*)d_in[2];
    const float* qp    = (const float*)d_in[3];
    const float* W_out = (const float*)d_in[4];
    const float* b_out = (const float*)d_in[5];
    float* out = (float*)d_out;

    // workspace carve-up: Wi16 (128KB) | We16 (128KB) | b_eff (4KB)
    _Float16* Wi16 = (_Float16*)d_ws;
    _Float16* We16 = (_Float16*)((char*)d_ws + 131072);
    float*    b_eff = (float*)((char*)d_ws + 262144);

    qnn_prep_win <<<(NQ * D_IN) / 256, 256, 0, stream>>>(W_in, Wi16);
    qnn_prep_wout<<<D_OUT / 256,       256, 0, stream>>>(qp, W_out, b_out, We16, b_eff);
    qnn_main     <<<BROWS / 128,       256, 0, stream>>>(x, b_in, Wi16, We16, b_eff, out);
}
